// OctreeResBlock2_33973191311682
// MI455X (gfx1250) — matrix operand order repaired and checked by reading, not compiled, not run on hardware
//
#include <hip/hip_runtime.h>
#include <hip/hip_bf16.h>
#include <stdint.h>
#include <stddef.h>

// ---------- CDNA5 WMMA vector types ----------
typedef __attribute__((ext_vector_type(16))) __bf16 v16bf;
typedef __attribute__((ext_vector_type(8)))  __bf16 v8bf;
typedef __attribute__((ext_vector_type(8)))  float  v8f;

#define COUT 256
#define NT   (COUT / 16)   // 16 column tiles of 16
#define KNB  27            // 3x3x3 octree kernel taps
#define EPSV 1e-5f

// ---------------------------------------------------------------------------
// fp32 -> bf16 cast (streaming)
// ---------------------------------------------------------------------------
__global__ void k_cast_bf16(const float* __restrict__ src,
                            __bf16* __restrict__ dst, long n) {
  long i = (long)blockIdx.x * blockDim.x + threadIdx.x;
  if (i < n) dst[i] = (__bf16)src[i];
}

// ---------------------------------------------------------------------------
// Pack W[ko][Cin][Cout] (f32 row-major) into per-WMMA-fragment bf16 layout.
// Fragment id = (ko*KC + kc)*NT + nt ; each fragment = 32 lanes x 16 elems
// stored contiguously so a lane loads its 32 bytes with 2x b128.
// B-fragment (32x16, bf16) lane map (per ISA 16-bit B layout):
//   lane L holds column n = L%16 ; elem j holds K = (L/16)*8 + (j%8) + (j/8)*16.
// ---------------------------------------------------------------------------
__global__ void k_pack_w(const float* __restrict__ W, __bf16* __restrict__ P,
                         int KO, int CIN) {
  const int KC = CIN / 32;
  const long total = (long)KO * KC * NT * 512;
  long t = (long)blockIdx.x * blockDim.x + threadIdx.x;
  if (t >= total) return;
  int  j    = (int)(t & 15);
  int  L    = (int)((t >> 4) & 31);
  long frag = t >> 9;
  int  nt   = (int)(frag % NT);
  int  kc   = (int)((frag / NT) % KC);
  int  ko   = (int)(frag / ((long)NT * KC));
  int  n    = L & 15;
  int  kk   = ((L >> 4) << 3) + (j & 7) + ((j >> 3) << 4);
  P[t] = (__bf16)W[((long)ko * CIN + kc * 32 + kk) * COUT + nt * 16 + n];
}

// ---------------------------------------------------------------------------
// Gather-GEMM with WMMA bf16 -> f32.
//   Y[m, :] = sum_ko  X[neigh[m,ko], :] @ Wko        (neigh==nullptr -> identity)
// Block = 256 threads = 8 wave32. Block owns 4 M-tiles (64 rows); wave w owns
// column tiles 2w and 2w+1 (Cout=256 covered once). B fragments are loaded
// once per (ko,kc) and reused from registers across the 4 M-tiles
// (1.5 vmem ops per WMMA). Next-tap gather rows are prefetched with
// global_prefetch_b8 to overlap random-gather latency with WMMA issue.
// A-fragment (16x32, bf16) lane map (per ISA 16-bit A layout):
//   lane L reads row M = L%16 ;
//   lanes 0-15  hold K = {0..7, 16..23}, lanes 16-31 hold K = {8..15, 24..31}
//   -> per lane two contiguous 16-byte loads per 32-wide K chunk.
// ---------------------------------------------------------------------------
template <int CIN>
__global__ void __launch_bounds__(256)
k_conv_wmma(const __bf16* __restrict__ X, const int* __restrict__ neigh,
            const __bf16* __restrict__ WP, float* __restrict__ Y,
            int N, int KO) {
  constexpr int KC = CIN / 32;
  constexpr int MT = 4;                       // M-tiles per block
  const int lane = threadIdx.x & 31;
  const int wave = threadIdx.x >> 5;
  const int half = lane >> 4;                 // 0: lanes 0-15, 1: lanes 16-31
  const int nt0  = wave * 2;

  int m[MT];
#pragma unroll
  for (int t = 0; t < MT; ++t) {
    int mm = blockIdx.x * (16 * MT) + t * 16 + (lane & 15);
    m[t] = (mm < N) ? mm : (N - 1);           // clamp; EXEC must stay all-1 for WMMA
  }

  v8f c[MT][2];
#pragma unroll
  for (int t = 0; t < MT; ++t) { c[t][0] = (v8f){}; c[t][1] = (v8f){}; }

  for (int ko = 0; ko < KO; ++ko) {
    const __bf16* __restrict__ ar[MT];
#pragma unroll
    for (int t = 0; t < MT; ++t) {
      const int r = neigh ? neigh[(long)m[t] * KNB + ko] : m[t];
      ar[t] = X + (long)r * CIN + (half << 3);
    }
    // Prefetch next tap's gathered rows (global_prefetch_b8) while this tap
    // computes: hides the random-gather HBM/L2 latency behind WMMA issue.
    if (neigh && (ko + 1 < KO)) {
#pragma unroll
      for (int t = 0; t < MT; ++t) {
        const int rn = neigh[(long)m[t] * KNB + (ko + 1)];
        const __bf16* pf = X + (long)rn * CIN;
        __builtin_prefetch(pf, 0, 3);
        if (CIN == 256) __builtin_prefetch(pf + 128, 0, 3);  // rows span 512B
      }
    }
    const __bf16* __restrict__ wf = WP + (long)ko * KC * NT * 512;
#pragma unroll
    for (int kc = 0; kc < KC; ++kc) {
      v16bf b0 = *(const v16bf*)(wf + ((long)(kc * NT + nt0)     * 32 + lane) * 16);
      v16bf b1 = *(const v16bf*)(wf + ((long)(kc * NT + nt0 + 1) * 32 + lane) * 16);
#pragma unroll
      for (int t = 0; t < MT; ++t) {
        v8bf alo = *(const v8bf*)(ar[t] + kc * 32);       // K base .. base+7
        v8bf ahi = *(const v8bf*)(ar[t] + kc * 32 + 16);  // K base+16 .. base+23
        v16bf a;
#pragma unroll
        for (int j = 0; j < 8; ++j) { a[j] = alo[j]; a[8 + j] = ahi[j]; }
        c[t][0] = __builtin_amdgcn_wmma_f32_16x16x32_bf16(false, a, false, b0,
                                                          (short)0, c[t][0], false, false);
        c[t][1] = __builtin_amdgcn_wmma_f32_16x16x32_bf16(false, a, false, b1,
                                                          (short)0, c[t][1], false, false);
      }
    }
  }

  // D layout: VGPR i -> M = half*8 + i ; N = lane%16 within the tile.
  const int col0 = nt0 * 16 + (lane & 15);
#pragma unroll
  for (int t = 0; t < MT; ++t) {
    const long rowbase = (long)blockIdx.x * (16 * MT) + t * 16 + half * 8;
#pragma unroll
    for (int i = 0; i < 8; ++i) {
      long row = rowbase + i;
      if (row < N) {
        Y[row * COUT + col0]      = c[t][0][i];
        Y[row * COUT + col0 + 16] = c[t][1][i];
      }
    }
  }
}

// ---------------------------------------------------------------------------
// Column sum / sum-of-squares over N rows (for BatchNorm training stats).
// Thread c owns channel c; coalesced grid-stride rows; one atomic per block.
// ---------------------------------------------------------------------------
__global__ void k_colstats(const float* __restrict__ Y, long N,
                           float* __restrict__ sum, float* __restrict__ sq) {
  const int c = threadIdx.x;  // 256 threads
  float s = 0.f, q = 0.f;
  for (long r = blockIdx.x; r < N; r += gridDim.x) {
    float v = Y[r * COUT + c];
    s += v;
    q += v * v;
  }
  atomicAdd(&sum[c], s);
  atomicAdd(&sq[c], q);
}

__global__ void k_zero(float* __restrict__ p, int n) {
  int i = blockIdx.x * blockDim.x + threadIdx.x;
  if (i < n) p[i] = 0.f;
}

// scale = gamma * rsqrt(var + eps) ; bias = beta - mean*scale
__global__ void k_finalize(const float* __restrict__ sum,
                           const float* __restrict__ sq,
                           const float* __restrict__ gamma,
                           const float* __restrict__ beta,
                           float* __restrict__ sb, float invN) {
  const int c = threadIdx.x;
  float m  = sum[c] * invN;
  float v  = sq[c] * invN - m * m;
  float sc = gamma[c] * rsqrtf(v + EPSV);
  sb[c]        = sc;
  sb[COUT + c] = beta[c] - m * sc;
}

// BN + ReLU, emit bf16 activation (conv2 gather source -> half the bytes)
__global__ void k_bn_relu_bf16(const float* __restrict__ Y,
                               const float* __restrict__ sb,
                               __bf16* __restrict__ out, long total) {
  long i = (long)blockIdx.x * blockDim.x + threadIdx.x;
  if (i >= total) return;
  int c = (int)(i & (COUT - 1));
  float v = fmaf(Y[i], sb[c], sb[COUT + c]);
  out[i] = (__bf16)fmaxf(v, 0.f);
}

// out = relu( bn(conv2) + bn(shortcut) )
__global__ void k_final(const float* __restrict__ Y2, const float* __restrict__ S,
                        const float* __restrict__ sb2, const float* __restrict__ sbs,
                        float* __restrict__ out, long total) {
  long i = (long)blockIdx.x * blockDim.x + threadIdx.x;
  if (i >= total) return;
  int c = (int)(i & (COUT - 1));
  float v = fmaf(Y2[i], sb2[c], sb2[COUT + c]) + fmaf(S[i], sbs[c], sbs[COUT + c]);
  out[i] = fmaxf(v, 0.f);
}

// ---------------------------------------------------------------------------
extern "C" void kernel_launch(void* const* d_in, const int* in_sizes, int n_in,
                              void* d_out, int out_size, void* d_ws, size_t ws_size,
                              hipStream_t stream) {
  const float* data  = (const float*)d_in[0];   // [N,128]
  const int*   neigh = (const int*)d_in[1];     // [N,27]
  // d_in[2] = depth (scalar, unused for stride==1 path)
  const float* Wa = (const float*)d_in[3];      // [27,128,256]
  const float* ga = (const float*)d_in[4];
  const float* ba = (const float*)d_in[5];
  const float* Wb = (const float*)d_in[6];      // [27,256,256]
  const float* gb = (const float*)d_in[7];
  const float* bb = (const float*)d_in[8];
  const float* W1 = (const float*)d_in[9];      // [128,256]
  const float* gs = (const float*)d_in[10];
  const float* bs = (const float*)d_in[11];
  float* out = (float*)d_out;

  const long N = (long)in_sizes[0] / 128;

  // ---- carve workspace (256B-aligned slabs) ----
  uint8_t* p = (uint8_t*)d_ws;
  auto carve = [&](size_t bytes) -> void* {
    void* q = (void*)p;
    p += (bytes + 255) & ~(size_t)255;
    return q;
  };
  __bf16* data_bf = (__bf16*)carve((size_t)N * 128 * sizeof(__bf16));
  float*  y1      = (float*) carve((size_t)N * COUT * sizeof(float));  // conv1 raw, later shortcut raw
  __bf16* y1act   = (__bf16*)carve((size_t)N * COUT * sizeof(__bf16));
  float*  y2      = (float*) carve((size_t)N * COUT * sizeof(float));
  __bf16* wa_p    = (__bf16*)carve((size_t)KNB * 4 * NT * 512 * sizeof(__bf16));
  __bf16* wb_p    = (__bf16*)carve((size_t)KNB * 8 * NT * 512 * sizeof(__bf16));
  __bf16* w1_p    = (__bf16*)carve((size_t)1   * 4 * NT * 512 * sizeof(__bf16));
  float*  stats   = (float*) carve((size_t)3072 * sizeof(float));
  float *sum1 = stats,        *sq1 = stats + 256;
  float *sum2 = stats + 512,  *sq2 = stats + 768;
  float *sum3 = stats + 1024, *sq3 = stats + 1280;
  float *sb1  = stats + 1536, *sb2 = stats + 2048, *sbs = stats + 2560;

  const long totA = N * 128;
  const long tot  = N * COUT;
  const int  mblocks = (int)((N + 63) / 64);   // 64 rows per block
  const float invN = 1.0f / (float)N;

  // ---- preprocessing ----
  k_cast_bf16<<<(int)((totA + 255) / 256), 256, 0, stream>>>(data, data_bf, totA);
  { long pa = (long)KNB * 4 * NT * 512;
    k_pack_w<<<(int)((pa + 255) / 256), 256, 0, stream>>>(Wa, wa_p, KNB, 128); }
  { long pb = (long)KNB * 8 * NT * 512;
    k_pack_w<<<(int)((pb + 255) / 256), 256, 0, stream>>>(Wb, wb_p, KNB, 256); }
  { long p1 = (long)1 * 4 * NT * 512;
    k_pack_w<<<(int)((p1 + 255) / 256), 256, 0, stream>>>(W1, w1_p, 1, 128); }
  k_zero<<<(3072 + 255) / 256, 256, 0, stream>>>(stats, 3072);

  // ---- conv1 (gather GEMM, bf16 WMMA) + BN stats + BN/ReLU -> bf16 ----
  k_conv_wmma<128><<<mblocks, 256, 0, stream>>>(data_bf, neigh, wa_p, y1, (int)N, KNB);
  k_colstats<<<1024, 256, 0, stream>>>(y1, N, sum1, sq1);
  k_finalize<<<1, 256, 0, stream>>>(sum1, sq1, ga, ba, sb1, invN);
  k_bn_relu_bf16<<<(int)((tot + 255) / 256), 256, 0, stream>>>(y1, sb1, y1act, tot);

  // ---- shortcut (reuses y1 slab) and conv2 ----
  k_conv_wmma<128><<<mblocks, 256, 0, stream>>>(data_bf, nullptr, w1_p, y1, (int)N, 1);
  k_conv_wmma<256><<<mblocks, 256, 0, stream>>>(y1act, neigh, wb_p, y2, (int)N, KNB);

  // ---- BN stats for conv2 + shortcut, fused epilogue ----
  k_colstats<<<1024, 256, 0, stream>>>(y2, N, sum2, sq2);
  k_colstats<<<1024, 256, 0, stream>>>(y1, N, sum3, sq3);
  k_finalize<<<1, 256, 0, stream>>>(sum2, sq2, gb, bb, sb2, invN);
  k_finalize<<<1, 256, 0, stream>>>(sum3, sq3, gs, bs, sbs, invN);
  k_final<<<(int)((tot + 255) / 256), 256, 0, stream>>>(y2, y1, sb2, sbs, out, tot);
}
